// RotaryMultiHeadAttention_32547262169468
// MI455X (gfx1250) — matrix operand order repaired and checked by reading, not compiled, not run on hardware
//
#include <hip/hip_runtime.h>
#include <hip/hip_bf16.h>
#include <math.h>

typedef __bf16 bf16;
typedef __attribute__((ext_vector_type(16))) __bf16 v16bf;
typedef __attribute__((ext_vector_type(8)))  __bf16 v8bf;
typedef __attribute__((ext_vector_type(8)))  float  v8f;
typedef __attribute__((ext_vector_type(4)))  float  v4f;

#define D_MODEL 1024
#define NHEAD   16
#define HEAD_DIM 64
#define SEQ     2048
#define BATCH   2
#define MTOT    (BATCH*SEQ)      // 4096

__device__ inline v8f wmma_bf16(v16bf a, v16bf b, v8f c) {
    // D = A(16x32 bf16) * B(32x16 bf16) + C(16x16 f32)
    return __builtin_amdgcn_wmma_f32_16x16x32_bf16(
        false, a, false, b, (short)0, c, false, false);
}

__device__ inline v16bf cat8(v8bf lo, v8bf hi) {
    v16bf r;
#pragma unroll
    for (int i = 0; i < 8; ++i) { r[i] = lo[i]; r[i + 8] = hi[i]; }
    return r;
}

// ---------------- f32 -> bf16 conversion (vectorized: 4 elems/thread) ----------------
__global__ void cvt_f32_bf16(const float* __restrict__ in, bf16* __restrict__ out, int n4) {
    int i = blockIdx.x * blockDim.x + threadIdx.x;
    if (i < n4) {
        v4f v = *(const v4f*)(in + (size_t)i * 4);
        bf16 o[4];
#pragma unroll
        for (int j = 0; j < 4; ++j) o[j] = (bf16)v[j];
        *(uint64_t*)(out + (size_t)i * 4) = *(const uint64_t*)o;
    }
}

// ---------------- WMMA GEMM: Y[4096,1024] = A[4096,1024] @ W[1024,1024]^T + bias
// One wave computes a 32x64 output tile (2 M-tiles x 4 N-tiles), K-loop step 32.
// mode 0: out bf16 in [b,h,s,d]   (Q, K)
// mode 1: out bf16 in [b,h,d,s]   (V transposed for attention B-fragments)
// mode 2: out f32  in [m,n]       (final projection -> d_out)
__global__ void gemm32_bf16(const bf16* __restrict__ A, const bf16* __restrict__ W,
                            const float* __restrict__ bias, void* __restrict__ outp, int mode) {
    const int lane = threadIdx.x;
    const int g    = lane >> 4;     // half-wave group
    const int l16  = lane & 15;
    const int mbase = blockIdx.x * 32;
    const int nbase = blockIdx.y * 64;

    v8f acc[2][4];
#pragma unroll
    for (int mt = 0; mt < 2; ++mt)
#pragma unroll
        for (int j = 0; j < 4; ++j)
#pragma unroll
            for (int r = 0; r < 8; ++r) acc[mt][j][r] = 0.0f;

    const bf16* arow0 = A + (size_t)(mbase + l16) * D_MODEL;
    const bf16* arow1 = A + (size_t)(mbase + 16 + l16) * D_MODEL;

    for (int kc = 0; kc < D_MODEL; kc += 32) {
        // A fragments: row = l16 (+16 for tile 1), k halves {8g..8g+7, 16+8g..16+8g+7}
        v16bf Af0 = cat8(*(const v8bf*)(arow0 + kc + 8 * g),
                         *(const v8bf*)(arow0 + kc + 16 + 8 * g));
        v16bf Af1 = cat8(*(const v8bf*)(arow1 + kc + 8 * g),
                         *(const v8bf*)(arow1 + kc + 16 + 8 * g));
#pragma unroll
        for (int j = 0; j < 4; ++j) {
            // B fragment: col n = nbase+16j+l16, k halves {8g..8g+7, 16+8g..16+8g+7}
            const bf16* wr = W + (size_t)(nbase + 16 * j + l16) * D_MODEL + kc;
            v16bf Bf = cat8(*(const v8bf*)(wr + 8 * g), *(const v8bf*)(wr + 16 + 8 * g));
            acc[0][j] = wmma_bf16(Af0, Bf, acc[0][j]);
            acc[1][j] = wmma_bf16(Af1, Bf, acc[1][j]);
        }
    }

#pragma unroll
    for (int mt = 0; mt < 2; ++mt)
#pragma unroll
        for (int j = 0; j < 4; ++j) {
            const int n = nbase + 16 * j + l16;
            const float bv = bias[n];
#pragma unroll
            for (int r = 0; r < 8; ++r) {
                const int m = mbase + 16 * mt + r + 8 * g;   // C layout: row = r + 8g
                const float val = acc[mt][j][r] + bv;
                if (mode == 2) {
                    ((float*)outp)[(size_t)m * D_MODEL + n] = val;
                } else {
                    const int b = m >> 11, s = m & (SEQ - 1);
                    const int h = n >> 6, d = n & (HEAD_DIM - 1);
                    bf16* ob = (bf16*)outp;
                    if (mode == 0)
                        ob[(((size_t)(b * NHEAD + h)) * SEQ + s) * HEAD_DIM + d] = (bf16)val;
                    else
                        ob[(((size_t)(b * NHEAD + h)) * HEAD_DIM + d) * SEQ + s] = (bf16)val;
                }
            }
        }
}

// ---------------- RoPE in place on [b*h, s, d] bf16 (scale folds 1/sqrt(d) for Q)
__global__ void rope_kernel(bf16* __restrict__ t, float scale) {
    int idx = blockIdx.x * blockDim.x + threadIdx.x;   // 2*16*2048*32 pairs
    int row = idx >> 5;           // [b*h*s]
    int j   = idx & 31;           // rotation pair index
    int s   = row & (SEQ - 1);
    size_t base = (size_t)row * HEAD_DIM;
    float inv = __powf(10000.0f, -(float)j * (1.0f / 32.0f));
    float ang = (float)s * inv;
    float si, co;
    __sincosf(ang, &si, &co);
    float x0 = (float)t[base + j];
    float x1 = (float)t[base + j + 32];
    t[base + j]      = (bf16)((x0 * co - x1 * si) * scale);
    t[base + j + 32] = (bf16)((x1 * co + x0 * si) * scale);
}

// ---------------- Flash attention: one wave per (b*h, 16-query tile)
// Q,K: [b*h, s, d] bf16 (Q pre-scaled by 1/8). Vt: [b*h, d, s] bf16.
// Out: [b, s, h*d] bf16.
__global__ void attn_kernel(const bf16* __restrict__ Q, const bf16* __restrict__ K,
                            const bf16* __restrict__ Vt, bf16* __restrict__ Out) {
    __shared__ bf16 sP[16 * 32];              // P tile staging C-layout -> A-layout
    const int bh    = blockIdx.y;
    const int qbase = blockIdx.x * 16;
    const int lane  = threadIdx.x;
    const int g     = lane >> 4;
    const int l16   = lane & 15;

    const bf16* Qp = Q  + (size_t)bh * SEQ * HEAD_DIM + (size_t)qbase * HEAD_DIM;
    const bf16* Kp = K  + (size_t)bh * SEQ * HEAD_DIM;
    const bf16* Vp = Vt + (size_t)bh * HEAD_DIM * SEQ;

    // Q A-fragments (d = 0..31 and 32..63), loaded once
    const bf16* qrow = Qp + (size_t)l16 * HEAD_DIM;
    v16bf qA0 = cat8(*(const v8bf*)(qrow + 8 * g),      *(const v8bf*)(qrow + 16 + 8 * g));
    v16bf qA1 = cat8(*(const v8bf*)(qrow + 32 + 8 * g), *(const v8bf*)(qrow + 48 + 8 * g));

    float m_run[8], l_run[8];
    v8f acc[4];
#pragma unroll
    for (int r = 0; r < 8; ++r) { m_run[r] = -1e30f; l_run[r] = 0.0f; }
#pragma unroll
    for (int tt = 0; tt < 4; ++tt)
#pragma unroll
        for (int r = 0; r < 8; ++r) acc[tt][r] = 0.0f;

    for (int kc = 0; kc < SEQ / 32; ++kc) {
        const int s0 = kc * 32;
        // ---- scores: S[16 x 32] = Q @ K^T, two 16-col tiles, two k=32 steps each
        v8f S[2];
#pragma unroll
        for (int j = 0; j < 2; ++j) {
            const bf16* kb = Kp + (size_t)(s0 + 16 * j + l16) * HEAD_DIM;
            v16bf B0 = cat8(*(const v8bf*)(kb + 8 * g), *(const v8bf*)(kb + 16 + 8 * g));        // d halves {8g.., 16+8g..}
            v16bf B1 = cat8(*(const v8bf*)(kb + 32 + 8 * g), *(const v8bf*)(kb + 48 + 8 * g));
            v8f c;
#pragma unroll
            for (int r = 0; r < 8; ++r) c[r] = 0.0f;
            c = wmma_bf16(qA0, B0, c);
            c = wmma_bf16(qA1, B1, c);
            S[j] = c;
        }

        __syncthreads();   // previous iteration's LDS reads done before overwrite
        // ---- online softmax (row = r + 8g, cols spread over 16 lanes of the half)
#pragma unroll
        for (int r = 0; r < 8; ++r) {
            float t = fmaxf(S[0][r], S[1][r]);
#pragma unroll
            for (int off = 8; off >= 1; off >>= 1)
                t = fmaxf(t, __shfl_xor(t, off, 32));
            const float mnew = fmaxf(m_run[r], t);
            const float sc   = __expf(m_run[r] - mnew);
            m_run[r] = mnew;
            const float p0 = __expf(S[0][r] - mnew);
            const float p1 = __expf(S[1][r] - mnew);
            l_run[r] = l_run[r] * sc + p0 + p1;     // per-lane partial; reduced at end
#pragma unroll
            for (int tt = 0; tt < 4; ++tt) acc[tt][r] *= sc;
            sP[(r + 8 * g) * 32 + l16]      = (bf16)p0;
            sP[(r + 8 * g) * 32 + 16 + l16] = (bf16)p1;
        }
        __syncthreads();

        // ---- P as A-fragment from LDS
        const bf16* lp = &sP[l16 * 32];
        v16bf pA = cat8(*(const v8bf*)(lp + 8 * g), *(const v8bf*)(lp + 16 + 8 * g));

        // ---- O += P @ V : Vt[d, s] gives contiguous B-fragment loads
#pragma unroll
        for (int tt = 0; tt < 4; ++tt) {
            const bf16* vb = Vp + (size_t)(16 * tt + l16) * SEQ + s0;
            v16bf vB = cat8(*(const v8bf*)(vb + 8 * g), *(const v8bf*)(vb + 16 + 8 * g));
            acc[tt] = wmma_bf16(pA, vB, acc[tt]);
        }
    }

    // final denominator reduction across the 16 lanes of each half
#pragma unroll
    for (int r = 0; r < 8; ++r)
#pragma unroll
        for (int off = 8; off >= 1; off >>= 1)
            l_run[r] += __shfl_xor(l_run[r], off, 32);

    const int b = bh >> 4, h = bh & 15;
#pragma unroll
    for (int tt = 0; tt < 4; ++tt)
#pragma unroll
        for (int r = 0; r < 8; ++r) {
            const int s = qbase + r + 8 * g;
            const float o = acc[tt][r] / l_run[r];
            Out[((size_t)b * SEQ + s) * D_MODEL + h * HEAD_DIM + 16 * tt + l16] = (bf16)o;
        }
}

extern "C" void kernel_launch(void* const* d_in, const int* in_sizes, int n_in,
                              void* d_out, int out_size, void* d_ws, size_t ws_size,
                              hipStream_t stream) {
    const float* x  = (const float*)d_in[0];
    const float* Wq = (const float*)d_in[1];
    const float* bq = (const float*)d_in[2];
    const float* Wk = (const float*)d_in[3];
    const float* bk = (const float*)d_in[4];
    const float* Wv = (const float*)d_in[5];
    const float* bv = (const float*)d_in[6];
    const float* Wo = (const float*)d_in[7];
    const float* bo = (const float*)d_in[8];

    char* ws = (char*)d_ws;
    const size_t MB = 1024 * 1024;
    bf16* xb    = (bf16*)(ws + 0);        // 8 MB  [4096,1024]
    bf16* Wqb   = (bf16*)(ws + 8  * MB);  // 2 MB
    bf16* Wkb   = (bf16*)(ws + 10 * MB);
    bf16* Wvb   = (bf16*)(ws + 12 * MB);
    bf16* Wob   = (bf16*)(ws + 14 * MB);
    bf16* qh    = (bf16*)(ws + 16 * MB);  // 8 MB [b,h,s,d]
    bf16* kh    = (bf16*)(ws + 24 * MB);  // 8 MB [b,h,s,d]
    bf16* vt    = (bf16*)(ws + 32 * MB);  // 8 MB [b,h,d,s]
    bf16* attnb = (bf16*)(ws + 40 * MB);  // 8 MB [b,s,1024]

    // 1) convert inputs to bf16 (4 elems per thread)
    cvt_f32_bf16<<<(MTOT * D_MODEL / 4 + 255) / 256, 256, 0, stream>>>(x, xb, MTOT * D_MODEL / 4);
    cvt_f32_bf16<<<(D_MODEL * D_MODEL / 4 + 255) / 256, 256, 0, stream>>>(Wq, Wqb, D_MODEL * D_MODEL / 4);
    cvt_f32_bf16<<<(D_MODEL * D_MODEL / 4 + 255) / 256, 256, 0, stream>>>(Wk, Wkb, D_MODEL * D_MODEL / 4);
    cvt_f32_bf16<<<(D_MODEL * D_MODEL / 4 + 255) / 256, 256, 0, stream>>>(Wv, Wvb, D_MODEL * D_MODEL / 4);
    cvt_f32_bf16<<<(D_MODEL * D_MODEL / 4 + 255) / 256, 256, 0, stream>>>(Wo, Wob, D_MODEL * D_MODEL / 4);

    // 2) QKV projections (WMMA, 32x64 tile per wave)
    dim3 ggrid(MTOT / 32, D_MODEL / 64);
    gemm32_bf16<<<ggrid, 32, 0, stream>>>(xb, Wqb, bq, qh, 0);
    gemm32_bf16<<<ggrid, 32, 0, stream>>>(xb, Wkb, bk, kh, 0);
    gemm32_bf16<<<ggrid, 32, 0, stream>>>(xb, Wvb, bv, vt, 1);

    // 3) RoPE (Q pre-scaled by 1/sqrt(head_dim) = 1/8)
    const int rope_threads = BATCH * NHEAD * SEQ * 32;
    rope_kernel<<<rope_threads / 256, 256, 0, stream>>>(qh, 0.125f);
    rope_kernel<<<rope_threads / 256, 256, 0, stream>>>(kh, 1.0f);

    // 4) flash attention (WMMA)
    attn_kernel<<<dim3(SEQ / 16, BATCH * NHEAD), 32, 0, stream>>>(qh, kh, vt, attnb);

    // 5) output projection -> f32 d_out
    gemm32_bf16<<<ggrid, 32, 0, stream>>>(attnb, Wob, bo, d_out, 2);
}
